// TemporalLocalAwareAttention_1632087573311
// MI455X (gfx1250) — compile-verified, matrix operand order checked by reading, not run on hardware
//
#include <hip/hip_runtime.h>

// ---------------------------------------------------------------------------
// TemporalLocalAwareAttention for MI455X (gfx1250, wave32, WMMA f16)
//   B=2, D=512, T=96, H=W=16  ->  N = 512 rows of (T=96, D=512)
//   nh=8, dh=64, local window s=3, factor = dh^-0.5 = 0.125
// ---------------------------------------------------------------------------

#define D_MODEL 512
#define T_SEQ   96
#define N_SEQ   512            // B*H*W
#define NH      8
#define DH      64
#define NROWS   (N_SEQ * T_SEQ) // 49152
#define FACTOR  0.125f
#define NEG_BIG -3.0e38f

typedef __attribute__((ext_vector_type(16))) _Float16 v16h;
typedef __attribute__((ext_vector_type(8)))  float    v8f;

// ---------------------------------------------------------------------------
// Fragment loader (ISA 7.12.2, 16-bit A-matrix 16x32 layout).
//   A-frag: source row-major [M,K], p -> element (0,0), leading dim ld.
//   B-frag: source [N,K] row-major (K stride 1): same addressing, "row" is the
//   N column. lanes 0-15 hold K=koff.., lanes 16-31 K=koff+8, vgprs 4-7 K+16.
// Compiler coalesces these into two ds_load_b128 per fragment.
// ---------------------------------------------------------------------------
__device__ __forceinline__ v16h load_frag(const _Float16* p, int ld) {
  const int lane = threadIdx.x & 31;
  const int row  = lane & 15;
  const int koff = (lane >> 4) << 3;          // 0 or 8
  const _Float16* r = p + row * ld + koff;
  v16h a;
#pragma unroll
  for (int i = 0; i < 4; ++i) {
    a[2*i]         = r[2*i];
    a[2*i + 1]     = r[2*i + 1];
    a[8 + 2*i]     = r[16 + 2*i];
    a[8 + 2*i + 1] = r[16 + 2*i + 1];
  }
  return a;
}

__device__ __forceinline__ v8f wmma_f16(v16h a, v16h b, v8f c) {
  return __builtin_amdgcn_wmma_f32_16x16x32_f16(false, a, false, b,
                                                (short)0, c, false, false);
}

// ---------------------------------------------------------------------------
// Kernel 1: LayerNorm over D, emit xn as f16 in [n*T + t, D] row-major.
// ---------------------------------------------------------------------------
__global__ __launch_bounds__(256) void ln_kernel(
    const float* __restrict__ x, const float* __restrict__ gamma,
    const float* __restrict__ beta, _Float16* __restrict__ xn) {
  const int bt = blockIdx.x;            // 0 .. 2*96-1
  const int b  = bt / T_SEQ;
  const int t  = bt % T_SEQ;
  const int hw = threadIdx.x;           // 0..255
  const size_t base    = ((size_t)b * D_MODEL * T_SEQ + t) * 256 + hw;
  const size_t dstride = (size_t)T_SEQ * 256;

  float sum = 0.f, sq = 0.f;
#pragma unroll 8
  for (int d = 0; d < D_MODEL; ++d) {
    float v = x[base + (size_t)d * dstride];
    sum += v; sq += v * v;
  }
  const float mu   = sum * (1.0f / D_MODEL);
  const float var  = sq  * (1.0f / D_MODEL) - mu * mu;
  const float rstd = rsqrtf(var + 1e-6f);

  _Float16* rowp = xn + ((size_t)(b * 256 + hw) * T_SEQ + t) * D_MODEL;
#pragma unroll 4
  for (int d = 0; d < D_MODEL; d += 2) {
    float v0 = x[base + (size_t)d * dstride];
    float v1 = x[base + (size_t)(d + 1) * dstride];
    rowp[d]     = (_Float16)((v0 - mu) * rstd * gamma[d]     + beta[d]);
    rowp[d + 1] = (_Float16)((v1 - mu) * rstd * gamma[d + 1] + beta[d + 1]);
  }
}

// ---------------------------------------------------------------------------
// Kernel 2: weights -> f16. Wcat = [Wq;Wk;Wv;Wkl;Wvl], stored [j,d] row-major
// (== [N,K] row-major for the B fragment loader). Wo separate.
// ---------------------------------------------------------------------------
__global__ __launch_bounds__(256) void wconv_kernel(
    const float* __restrict__ Wq, const float* __restrict__ Wk,
    const float* __restrict__ Wv, const float* __restrict__ Wkl,
    const float* __restrict__ Wvl, const float* __restrict__ Wo,
    _Float16* __restrict__ Wcat, _Float16* __restrict__ Wo16) {
  const int per = D_MODEL * D_MODEL;
  int idx = blockIdx.x * blockDim.x + threadIdx.x;
  if (idx >= per * 6) return;
  int m = idx / per, off = idx % per;
  float v;
  if      (m == 0) v = Wq[off];
  else if (m == 1) v = Wk[off];
  else if (m == 2) v = Wv[off];
  else if (m == 3) v = Wkl[off];
  else if (m == 4) v = Wvl[off];
  else             v = Wo[off];
  if (m < 5) Wcat[(size_t)m * per + off] = (_Float16)v;
  else       Wo16[off] = (_Float16)v;
}

// ---------------------------------------------------------------------------
// Pipelined WMMA GEMM core: 128x64 block tile, K-step 64, double-buffered LDS,
// 256 threads = 8 waves in a 4(M) x 2(N) grid, each wave owns a 32x32 patch
// (2x2 subtiles, 4 f32 accumulators). Per K-chunk of 32: 4 frag loads ->
// 4 WMMAs (1:1 LDS:WMMA ratio). One barrier per K-step.
// ---------------------------------------------------------------------------
#define LDT 80   // LDS leading dim in halves (160B rows: 16B aligned, skewed)

#define GEMM_PIPELINE_BODY(APTR, BPTR)                                         \
  __shared__ _Float16 As[2][128 * LDT];                                        \
  __shared__ _Float16 Bs[2][64 * LDT];                                         \
  const int tid  = threadIdx.x;                                                \
  const int wave = tid >> 5;                                                   \
  const int lane = tid & 31;                                                   \
  const int mBase = blockIdx.y * 128;                                          \
  const int nBase = blockIdx.x * 64;                                           \
  const int m0 = (wave >> 1) * 32;       /* wave's 32-row patch in tile  */    \
  const int n0 = (wave & 1) * 32;        /* wave's 32-col patch in tile  */    \
  v8f acc00 = {}, acc01 = {}, acc10 = {}, acc11 = {};                          \
  uint4 ra[4], rb[2];                                                          \
  /* stage K-tile 0 */                                                         \
  _Pragma("unroll")                                                            \
  for (int i = 0; i < 4; ++i) {                                                \
    int idx = tid + i * 256, r = idx >> 3, ch = idx & 7;                       \
    ra[i] = *(const uint4*)(APTR + (size_t)(mBase + r) * D_MODEL + ch * 8);    \
  }                                                                            \
  _Pragma("unroll")                                                            \
  for (int i = 0; i < 2; ++i) {                                                \
    int idx = tid + i * 256, r = idx >> 3, ch = idx & 7;                       \
    rb[i] = *(const uint4*)(BPTR + (size_t)(nBase + r) * D_MODEL + ch * 8);    \
  }                                                                            \
  _Pragma("unroll")                                                            \
  for (int i = 0; i < 4; ++i) {                                                \
    int idx = tid + i * 256, r = idx >> 3, ch = idx & 7;                       \
    *(uint4*)(&As[0][r * LDT + ch * 8]) = ra[i];                               \
  }                                                                            \
  _Pragma("unroll")                                                            \
  for (int i = 0; i < 2; ++i) {                                                \
    int idx = tid + i * 256, r = idx >> 3, ch = idx & 7;                       \
    *(uint4*)(&Bs[0][r * LDT + ch * 8]) = rb[i];                               \
  }                                                                            \
  __syncthreads();                                                             \
  for (int kk = 0; kk < 8; ++kk) {                                             \
    const int cur = kk & 1, nxt = cur ^ 1;                                     \
    if (kk < 7) { /* issue global loads for next K-tile (hidden by WMMAs) */   \
      _Pragma("unroll")                                                        \
      for (int i = 0; i < 4; ++i) {                                            \
        int idx = tid + i * 256, r = idx >> 3, ch = idx & 7;                   \
        ra[i] = *(const uint4*)(APTR + (size_t)(mBase + r) * D_MODEL +         \
                                (kk + 1) * 64 + ch * 8);                       \
      }                                                                        \
      _Pragma("unroll")                                                        \
      for (int i = 0; i < 2; ++i) {                                            \
        int idx = tid + i * 256, r = idx >> 3, ch = idx & 7;                   \
        rb[i] = *(const uint4*)(BPTR + (size_t)(nBase + r) * D_MODEL +         \
                                (kk + 1) * 64 + ch * 8);                       \
      }                                                                        \
      if (kk < 6) {                                                            \
        __builtin_prefetch(APTR + (size_t)(mBase + (tid >> 1)) * D_MODEL +     \
                           (kk + 2) * 64, 0, 3);                               \
      }                                                                        \
    }                                                                          \
    _Pragma("unroll")                                                          \
    for (int kc = 0; kc < 2; ++kc) {                                           \
      v16h aR0 = load_frag(&As[cur][m0 * LDT + kc * 32],        LDT);          \
      v16h aR1 = load_frag(&As[cur][(m0 + 16) * LDT + kc * 32], LDT);          \
      v16h bC0 = load_frag(&Bs[cur][n0 * LDT + kc * 32],        LDT);          \
      v16h bC1 = load_frag(&Bs[cur][(n0 + 16) * LDT + kc * 32], LDT);          \
      acc00 = wmma_f16(aR0, bC0, acc00);                                       \
      acc01 = wmma_f16(aR0, bC1, acc01);                                       \
      acc10 = wmma_f16(aR1, bC0, acc10);                                       \
      acc11 = wmma_f16(aR1, bC1, acc11);                                       \
    }                                                                          \
    if (kk < 7) {                                                              \
      _Pragma("unroll")                                                        \
      for (int i = 0; i < 4; ++i) {                                            \
        int idx = tid + i * 256, r = idx >> 3, ch = idx & 7;                   \
        *(uint4*)(&As[nxt][r * LDT + ch * 8]) = ra[i];                         \
      }                                                                        \
      _Pragma("unroll")                                                        \
      for (int i = 0; i < 2; ++i) {                                            \
        int idx = tid + i * 256, r = idx >> 3, ch = idx & 7;                   \
        *(uint4*)(&Bs[nxt][r * LDT + ch * 8]) = rb[i];                         \
      }                                                                        \
    }                                                                          \
    __syncthreads();                                                           \
  }                                                                            \
  const int colL = lane & 15;                                                  \
  const int rowO = (lane >> 4) * 8;

// ---------------------------------------------------------------------------
// Kernel 3: projection GEMM. Y[r, j] = sum_d xn[r,d] * Wcat[j,d]
//   M = 49152, N = 2560, K = 512. Epilogue scatters q/k/kl/vl [n,h,t,dh] and
//   vT [n,h,dh,t] (f16), with q_bias + attention scale folded into q.
// ---------------------------------------------------------------------------
__global__ __launch_bounds__(256) void proj_gemm_kernel(
    const _Float16* __restrict__ xn, const _Float16* __restrict__ Wcat,
    const float* __restrict__ q_bias,
    _Float16* __restrict__ qb, _Float16* __restrict__ kb,
    _Float16* __restrict__ vTb, _Float16* __restrict__ klb,
    _Float16* __restrict__ vlb) {
  GEMM_PIPELINE_BODY(xn, Wcat)
#pragma unroll
  for (int si = 0; si < 2; ++si) {
#pragma unroll
    for (int sj = 0; sj < 2; ++sj) {
      v8f acc = (si == 0) ? (sj == 0 ? acc00 : acc01)
                          : (sj == 0 ? acc10 : acc11);
      int j    = nBase + n0 + sj * 16 + colL;
      int mat  = j >> 9;                // 0=q 1=k 2=v 3=kl 4=vl
      int jj   = j & 511;
      int head = jj >> 6, dd = jj & 63;
#pragma unroll
      for (int i = 0; i < 8; ++i) {
        int r = mBase + m0 + si * 16 + rowO + i;
        int n = r / T_SEQ, t = r % T_SEQ;
        float v = acc[i];
        size_t ohd = ((size_t)(n * NH + head) * T_SEQ + t) * DH + dd;
        if (mat == 0) {
          v = (v + q_bias[jj]) * FACTOR;   // fold bias + scale into q
          qb[ohd] = (_Float16)v;
        } else if (mat == 1) {
          kb[ohd] = (_Float16)v;
        } else if (mat == 2) {
          vTb[((size_t)(n * NH + head) * DH + dd) * T_SEQ + t] = (_Float16)v;
        } else if (mat == 3) {
          klb[ohd] = (_Float16)v;
        } else {
          vlb[ohd] = (_Float16)v;
        }
      }
    }
  }
}

// ---------------------------------------------------------------------------
// Kernel 4: attention, one block per (n, h). 128 threads = 4 waves.
// LDS: q,k,kl,vl [96][64] f16; vT [64][96] f16; scores [96][100] f32
//      (96 global + 3 local); probs P [96][96] f16; local probs [96][4] f32.
// ---------------------------------------------------------------------------
#define SLD 100

__global__ __launch_bounds__(128) void attn_kernel(
    const _Float16* __restrict__ qg,  const _Float16* __restrict__ kg,
    const _Float16* __restrict__ vTg, const _Float16* __restrict__ klg,
    const _Float16* __restrict__ vlg, _Float16* __restrict__ att_out) {
  extern __shared__ char smem[];
  _Float16* q_s  = (_Float16*)smem;            // 96*64
  _Float16* k_s  = q_s  + 96 * 64;
  _Float16* kl_s = k_s  + 96 * 64;
  _Float16* vl_s = kl_s + 96 * 64;
  _Float16* vT_s = vl_s + 96 * 64;             // 64*96
  float*    S_s  = (float*)(vT_s + 64 * 96);   // 96*SLD f32
  _Float16* P_s  = (_Float16*)(S_s + 96 * SLD);// 96*96 f16
  float*    AL_s = (float*)(P_s + 96 * 96);    // 96*4 f32

  const int nh_id = blockIdx.x;                // n*8 + h
  const int tid   = threadIdx.x;
  const int wave  = tid >> 5;
  const int lane  = tid & 31;

  { // stage all operands into LDS (each tile is 768 uint4)
    const size_t base = (size_t)nh_id * (T_SEQ * DH);
    const uint4* s0 = (const uint4*)(qg  + base);
    const uint4* s1 = (const uint4*)(kg  + base);
    const uint4* s2 = (const uint4*)(klg + base);
    const uint4* s3 = (const uint4*)(vlg + base);
    const uint4* s4 = (const uint4*)(vTg + base);
    uint4* d0 = (uint4*)q_s;  uint4* d1 = (uint4*)k_s;
    uint4* d2 = (uint4*)kl_s; uint4* d3 = (uint4*)vl_s;
    uint4* d4 = (uint4*)vT_s;
    for (int i = tid; i < 768; i += 128) {
      d0[i] = s0[i]; d1[i] = s1[i]; d2[i] = s2[i]; d3[i] = s3[i]; d4[i] = s4[i];
    }
  }
  __syncthreads();

  // ---- global scores: S = q @ k^T  (scale already folded into q) ----
  for (int job = wave; job < 36; job += 4) {
    int rt = job / 6, ct = job % 6;
    v8f c = {};
    v16h a0 = load_frag(q_s + rt * 16 * DH,      DH);
    v16h a1 = load_frag(q_s + rt * 16 * DH + 32, DH);
    v16h b0 = load_frag(k_s + ct * 16 * DH,      DH);
    v16h b1 = load_frag(k_s + ct * 16 * DH + 32, DH);
    c = wmma_f16(a0, b0, c);
    c = wmma_f16(a1, b1, c);
    int colS = ct * 16 + (lane & 15);
    int rb   = rt * 16 + ((lane >> 4) * 8);
#pragma unroll
    for (int i = 0; i < 8; ++i) S_s[(rb + i) * SLD + colS] = c[i];
  }
  // ---- local (3-wide window) scores via VALU ----
  if (tid < T_SEQ) {
    int t = tid;
#pragma unroll
    for (int l = 0; l < 3; ++l) {
      int tk = t + l - 1;
      float sc = NEG_BIG;                      // masked edge -> -inf
      if (tk >= 0 && tk < T_SEQ) {
        float a = 0.f;
        for (int d = 0; d < DH; ++d)
          a += (float)q_s[t * DH + d] * (float)kl_s[tk * DH + d];
        sc = a;
      }
      S_s[t * SLD + T_SEQ + l] = sc;
    }
  }
  __syncthreads();

  // ---- softmax over 99 entries per row ----
  if (tid < T_SEQ) {
    int t = tid;
    float m = NEG_BIG;
    for (int g = 0; g < T_SEQ + 3; ++g) m = fmaxf(m, S_s[t * SLD + g]);
    float sum = 0.f;
    for (int g = 0; g < T_SEQ + 3; ++g) sum += __expf(S_s[t * SLD + g] - m);
    float inv = 1.0f / sum;
    for (int g = 0; g < T_SEQ; ++g)
      P_s[t * T_SEQ + g] = (_Float16)(__expf(S_s[t * SLD + g] - m) * inv);
#pragma unroll
    for (int l = 0; l < 3; ++l)
      AL_s[t * 4 + l] = __expf(S_s[t * SLD + T_SEQ + l] - m) * inv;
  }
  __syncthreads();

  // ---- x_global = P @ V (WMMA), + local window contribution in epilogue ----
  const int n_id = nh_id >> 3, h_id = nh_id & 7;
  for (int job = wave; job < 24; job += 4) {
    int rt = job >> 2, ct = job & 3;
    v8f c = {};
#pragma unroll
    for (int kc = 0; kc < 3; ++kc) {           // K = 96 = 3 x 32
      v16h a = load_frag(P_s  + rt * 16 * T_SEQ + kc * 32, T_SEQ);
      v16h b = load_frag(vT_s + ct * 16 * T_SEQ + kc * 32, T_SEQ);
      c = wmma_f16(a, b, c);
    }
    int dcol = ct * 16 + (lane & 15);
    int rb   = rt * 16 + ((lane >> 4) * 8);
#pragma unroll
    for (int i = 0; i < 8; ++i) {
      int t = rb + i;
      float xl = 0.f;
#pragma unroll
      for (int l = 0; l < 3; ++l) {
        int tk = t + l - 1;
        if (tk >= 0 && tk < T_SEQ)
          xl += AL_s[t * 4 + l] * (float)vl_s[tk * DH + dcol];
      }
      float val = c[i] + xl;
      att_out[((size_t)(n_id * T_SEQ + t)) * D_MODEL + h_id * DH + dcol] =
          (_Float16)val;
    }
  }
}

// ---------------------------------------------------------------------------
// Kernel 5: output GEMM vs Wo + final permutation to (B, D, T, H, W), f32.
//   out[r, j] = att[r, :] . Wo[j, :],  r = ((b*16+h)*16+w)*96 + t
// ---------------------------------------------------------------------------
__global__ __launch_bounds__(256) void out_gemm_kernel(
    const _Float16* __restrict__ A, const _Float16* __restrict__ Wo16,
    float* __restrict__ y) {
  GEMM_PIPELINE_BODY(A, Wo16)
#pragma unroll
  for (int si = 0; si < 2; ++si) {
#pragma unroll
    for (int sj = 0; sj < 2; ++sj) {
      v8f acc = (si == 0) ? (sj == 0 ? acc00 : acc01)
                          : (sj == 0 ? acc10 : acc11);
      int j = nBase + n0 + sj * 16 + colL;    // output channel d
#pragma unroll
      for (int i = 0; i < 8; ++i) {
        int r = mBase + m0 + si * 16 + rowO + i;
        int n = r / T_SEQ, t = r % T_SEQ;
        int b = n >> 8, hh = (n >> 4) & 15, w = n & 15;
        // y[b, j, t, hh, w]
        y[(((size_t)(b * D_MODEL + j) * T_SEQ + t) * 16 + hh) * 16 + w] = acc[i];
      }
    }
  }
}

// ---------------------------------------------------------------------------
// Host launcher
// ---------------------------------------------------------------------------
extern "C" void kernel_launch(void* const* d_in, const int* in_sizes, int n_in,
                              void* d_out, int out_size, void* d_ws, size_t ws_size,
                              hipStream_t stream) {
  const float* x      = (const float*)d_in[0];
  const float* gammap = (const float*)d_in[1];
  const float* betap  = (const float*)d_in[2];
  const float* Wq     = (const float*)d_in[3];
  const float* Wk     = (const float*)d_in[4];
  const float* Wv     = (const float*)d_in[5];
  const float* Wkl    = (const float*)d_in[6];
  const float* Wvl    = (const float*)d_in[7];
  const float* Wo     = (const float*)d_in[8];
  const float* q_bias = (const float*)d_in[9];
  float* y = (float*)d_out;

  // workspace carve (bytes, 256B aligned)
  char* ws = (char*)d_ws;
  size_t off = 0;
  auto carve = [&](size_t bytes) -> void* {
    void* p = (void*)(ws + off);
    off = (off + bytes + 255) & ~(size_t)255;
    return p;
  };
  const size_t rowsBytes = (size_t)NROWS * D_MODEL * sizeof(_Float16);   // 50.3 MB
  const size_t projBytes = (size_t)N_SEQ * NH * T_SEQ * DH * sizeof(_Float16);

  _Float16* xn    = (_Float16*)carve(rowsBytes);   // reused as att_out later
  _Float16* Wcat  = (_Float16*)carve((size_t)5 * D_MODEL * D_MODEL * sizeof(_Float16));
  _Float16* Wo16  = (_Float16*)carve((size_t)D_MODEL * D_MODEL * sizeof(_Float16));
  _Float16* qb    = (_Float16*)carve(projBytes);
  _Float16* kb    = (_Float16*)carve(projBytes);
  _Float16* vTb   = (_Float16*)carve(projBytes);
  _Float16* klb   = (_Float16*)carve(projBytes);
  _Float16* vlb   = (_Float16*)carve(projBytes);
  (void)ws_size; (void)in_sizes; (void)n_in; (void)out_size;

  // 1) layernorm -> xn (f16)
  ln_kernel<<<dim3(2 * T_SEQ), dim3(256), 0, stream>>>(x, gammap, betap, xn);

  // 2) weights -> f16
  {
    int total = D_MODEL * D_MODEL * 6;
    wconv_kernel<<<dim3((total + 255) / 256), dim3(256), 0, stream>>>(
        Wq, Wk, Wv, Wkl, Wvl, Wo, Wcat, Wo16);
  }

  // 3) fused 5-way projection GEMM (M=49152, N=2560, K=512), 128x64 tiles
  proj_gemm_kernel<<<dim3(40, 384), dim3(256), 0, stream>>>(
      xn, Wcat, q_bias, qb, kb, vTb, klb, vlb);

  // 4) attention per (n, h); att_out reuses the xn buffer ([N*T, D] f16)
  {
    size_t smem = (size_t)(4 * 96 * 64 + 64 * 96) * sizeof(_Float16)  // q,k,kl,vl,vT
                + (size_t)96 * SLD * sizeof(float)                    // scores
                + (size_t)96 * 96 * sizeof(_Float16)                  // probs
                + (size_t)96 * 4 * sizeof(float);                     // local probs
    attn_kernel<<<dim3(N_SEQ * NH), dim3(128), smem, stream>>>(
        qb, kb, vTb, klb, vlb, xn);
  }

  // 5) output projection + final (B,D,T,H,W) permutation, 128x64 tiles
  out_gemm_kernel<<<dim3(8, 384), dim3(256), 0, stream>>>(xn, Wo16, y);
}
